// LSTMCell_43800076484839
// MI455X (gfx1250) — compile-verified, matrix operand order checked by reading, not run on hardware
//
#include <hip/hip_runtime.h>
#include <hip/hip_bf16.h>
#include <math.h>

typedef __attribute__((ext_vector_type(8)))  _Float16 v8h;
typedef __attribute__((ext_vector_type(16))) _Float16 v16h;
typedef __attribute__((ext_vector_type(8)))  float    v8f;

#define B_DIM 4096
#define I_DIM 1024
#define H_DIM 1024
#define K_DIM 2048      // I + H concatenated reduction dim
#define NG    4096      // 4*H gate columns
#define NLVL  0.05f

// GEMM tiling
#define BM   128
#define BH   32
#define KT   32
#define NIT  (K_DIM / KT)     // 64
#define XSTR 40               // halfs per LDS row (padded, 80B -> conflict-free)
#define WSTR 40
#define XBUF (BM * XSTR)      // halfs per X buffer
#define WBUF (4 * BH * WSTR)  // halfs per W buffer

#define FRAG16(p) __builtin_shufflevector(*(const v8h*)(p), *(const v8h*)((p) + 16), \
                                          0,1,2,3,4,5,6,7,8,9,10,11,12,13,14,15)

// Split barrier draining LDS ops AND async global->LDS DMA (CDNA5 ASYNCcnt).
#define ASYNC_LDS_BARRIER() asm volatile("s_wait_dscnt 0x0\n\t"    \
                                         "s_wait_asynccnt 0x0\n\t" \
                                         "s_barrier_signal -1\n\t" \
                                         "s_barrier_wait -1" ::: "memory")

// CDNA5 async memory->LDS copy (GLOBAL_LOAD_ASYNC_TO_LDS_B128, ASYNCcnt-tracked):
// per-lane 16B from global addr -> LDS byte offset, no VGPR staging.
__device__ __forceinline__ void async_ld_b128(unsigned lds_off, const void* gaddr) {
    asm volatile("global_load_async_to_lds_b128 %0, %1, off"
                 :: "v"(lds_off), "v"(gaddr)
                 : "memory");
}

// ---------- order-preserving float<->uint keys for atomic max ----------
__device__ __forceinline__ unsigned f2key(float f) {
    unsigned u = __float_as_uint(f);
    return (u & 0x80000000u) ? ~u : (u | 0x80000000u);
}
__device__ __forceinline__ float key2f(unsigned k) {
    return (k & 0x80000000u) ? __uint_as_float(k & 0x7FFFFFFFu)
                             : __uint_as_float(~k);
}

__global__ void init_max_kernel(unsigned* m) {
    if (threadIdx.x < 4) m[threadIdx.x] = 0u;  // key 0 == -inf-ish identity
}

__global__ void reduce_max_kernel(const float* __restrict__ src, int n, unsigned* out) {
    __shared__ float smax[256];
    float m = -INFINITY;
    for (int i = blockIdx.x * blockDim.x + threadIdx.x; i < n; i += gridDim.x * blockDim.x)
        m = fmaxf(m, src[i]);
    smax[threadIdx.x] = m;
    __syncthreads();
    for (int s = 128; s > 0; s >>= 1) {
        if (threadIdx.x < s) smax[threadIdx.x] = fmaxf(smax[threadIdx.x], smax[threadIdx.x + s]);
        __syncthreads();
    }
    if (threadIdx.x == 0) atomicMax(out, f2key(smax[0]));
}

// ---------- quant_pass (bits=8), round-to-nearest-even like jnp.round ----------
__device__ __forceinline__ float quantS8(float x, int train) {
    const float delta = 1.0f / 128.0f;
    float xc = fminf(fmaxf(x, -1.0f + delta), 1.0f - delta);
    float q  = rintf(xc * 128.0f) * (1.0f / 128.0f);
    return train ? 0.5f * q + 0.5f * xc : q;
}
__device__ __forceinline__ float quantU8(float x, int train) {
    float xc = fminf(fmaxf(x, 0.0f), 1.0f);
    float q  = rintf(xc * 256.0f) * (1.0f / 256.0f);
    return train ? 0.5f * q + 0.5f * xc : q;
}

// ---------- combined bias: b_ih + nb_ih + b_hh + nb_hh ----------
__global__ void bias_prep_kernel(const float* __restrict__ b_ih, const float* __restrict__ b_hh,
                                 const float* __restrict__ nb_ih, const float* __restrict__ nb_hh,
                                 const unsigned* __restrict__ maxes, float* __restrict__ biasC) {
    int n = blockIdx.x * blockDim.x + threadIdx.x;
    if (n >= NG) return;
    float mb_ih = key2f(maxes[2]);
    float mb_hh = key2f(maxes[3]);
    biasC[n] = b_ih[n] + nb_ih[n] * mb_ih * NLVL + b_hh[n] + nb_hh[n] * mb_hh * NLVL;
}

// ---------- Weff[n][k] f16 = W[n][k] + max(W)*NL*noise[k][n]; LDS transpose for noise ----------
__global__ void weight_prep_kernel(const float* __restrict__ w_ih, const float* __restrict__ w_hh,
                                   const float* __restrict__ nw_ih, const float* __restrict__ nw_hh,
                                   const unsigned* __restrict__ maxes, _Float16* __restrict__ Weff) {
    __shared__ float nts[32][33];
    int k0 = blockIdx.x * 32;   // 0..2047
    int n0 = blockIdx.y * 32;   // 0..4095
    int tx = threadIdx.x, ty = threadIdx.y;   // (32,8)
    bool ih = (k0 < I_DIM);
    const float* W  = ih ? w_ih  : w_hh;      // [NG][1024] row-major
    const float* NW = ih ? nw_ih : nw_hh;     // [1024][NG] row-major (transposed)
    int kk0 = ih ? k0 : (k0 - I_DIM);
    float scale = key2f(maxes[ih ? 0 : 1]) * NLVL;
#pragma unroll
    for (int i = 0; i < 4; i++) {
        int k = ty * 4 + i;
        nts[k][tx] = NW[(size_t)(kk0 + k) * NG + (n0 + tx)];   // coalesced in tx
    }
    __syncthreads();
#pragma unroll
    for (int i = 0; i < 4; i++) {
        int n = ty * 4 + i;
        float w = W[(size_t)(n0 + n) * I_DIM + (kk0 + tx)];    // coalesced in tx
        Weff[(size_t)(n0 + n) * K_DIM + (k0 + tx)] = (_Float16)(w + nts[tx][n] * scale);
    }
}

// ---------- Xq[b][k] f16 = quantS8(input | hx) ----------
__global__ void x_prep_kernel(const float* __restrict__ inp, const float* __restrict__ hx,
                              const int* __restrict__ trainp, _Float16* __restrict__ Xq) {
    int idx = blockIdx.x * blockDim.x + threadIdx.x;  // B*K
    if (idx >= B_DIM * K_DIM) return;
    int b = idx / K_DIM, k = idx - b * K_DIM;
    int train = trainp[0];
    float x = (k < I_DIM) ? inp[(size_t)b * I_DIM + k]
                          : hx[(size_t)b * H_DIM + (k - I_DIM)];
    Xq[idx] = (_Float16)quantS8(x, train);
}

// ---------- fused gates GEMM (f16 WMMA, f32 acc, async double-buffered LDS) + LSTM epilogue ----------
__global__ __launch_bounds__(256) void lstm_gemm_kernel(
    const _Float16* __restrict__ Xq,    // [B][K] f16
    const _Float16* __restrict__ Weff,  // [NG][K] f16
    const float* __restrict__ biasC,    // [NG]
    const float* __restrict__ cx,       // [B][H]
    const int* __restrict__ trainp,
    float* __restrict__ hy, float* __restrict__ cy)
{
    __shared__ __align__(16) _Float16 Xs[2 * XBUF];
    __shared__ __align__(16) _Float16 Ws[2 * WBUF];

    const int tid   = threadIdx.x;
    const int lane  = tid & 31;
    const int wave  = tid >> 5;
    const int waveM = wave & 3;     // 4 M-groups of 32 rows
    const int waveH = wave >> 2;    // 2 H-groups of 16 cols
    const int m0 = blockIdx.x * BM;
    const int h0 = blockIdx.y * BH;

    const int half = lane >> 4;     // ISA frag layout: lanes 16-31 hold K+8 halves
    const int l16  = lane & 15;

    v8f acc[4][2];
#pragma unroll
    for (int g = 0; g < 4; ++g) { acc[g][0] = (v8f){}; acc[g][1] = (v8f){}; }

    // ---- per-thread global source addresses (advance 64B per K-step) ----
    const int xr = tid >> 2, xc = tid & 3;          // X: rows xr and xr+64
    const _Float16* pX0 = &Xq[(size_t)(m0 + xr) * K_DIM + xc * 8];
    const _Float16* pX1 = pX0 + (size_t)64 * K_DIM;
    const int g0 = tid >> 7,         wr0 = (tid >> 2) & 31,          wc0 = tid & 3;
    const int g1 = (tid + 256) >> 7, wr1 = ((tid + 256) >> 2) & 31,  wc1 = tid & 3;
    const _Float16* pW0 = &Weff[(size_t)(h0 + g0 * 1024 + wr0) * K_DIM + wc0 * 8];
    const _Float16* pW1 = &Weff[(size_t)(h0 + g1 * 1024 + wr1) * K_DIM + wc1 * 8];

    // ---- per-thread LDS destination offsets (buffer 0; buffer 1 adds byte size) ----
    const unsigned ldsX0 = (unsigned)(size_t)&Xs[xr * XSTR + xc * 8];
    const unsigned ldsX1 = (unsigned)(size_t)&Xs[(xr + 64) * XSTR + xc * 8];
    const unsigned ldsW0 = (unsigned)(size_t)&Ws[(g0 * BH + wr0) * WSTR + wc0 * 8];
    const unsigned ldsW1 = (unsigned)(size_t)&Ws[(g1 * BH + wr1) * WSTR + wc1 * 8];
    const unsigned xbb = XBUF * sizeof(_Float16);   // X buffer byte stride
    const unsigned wbb = WBUF * sizeof(_Float16);   // W buffer byte stride

    const int aoff = (waveM * 32 + l16) * XSTR + half * 8;   // A frag base (row group 0)
    const int boff = (waveH * 16 + l16) * WSTR + half * 8;   // B frag base (gate 0)

    // ---- prologue: async DMA of tile 0 into buffer 0 ----
    async_ld_b128(ldsX0, pX0);
    async_ld_b128(ldsX1, pX1);
    async_ld_b128(ldsW0, pW0);
    async_ld_b128(ldsW1, pW1);
    pX0 += KT; pX1 += KT; pW0 += KT; pW1 += KT;

#pragma unroll 2
    for (int it = 0; it < NIT; ++it) {
        const int cur = it & 1;

        ASYNC_LDS_BARRIER();   // buffer[cur] DMA complete across the workgroup

        // kick off async DMA of tile it+1 into the other buffer; it has the
        // whole WMMA block (and next barrier) to complete in the background
        if (it + 1 < NIT) {
            const unsigned nb = (unsigned)(cur ^ 1);
            async_ld_b128(ldsX0 + nb * xbb, pX0);
            async_ld_b128(ldsX1 + nb * xbb, pX1);
            async_ld_b128(ldsW0 + nb * wbb, pW0);
            async_ld_b128(ldsW1 + nb * wbb, pW1);
            pX0 += KT; pX1 += KT; pW0 += KT; pW1 += KT;
        }

        const _Float16* xb = &Xs[cur * XBUF];
        const _Float16* wb = &Ws[cur * WBUF];
        const _Float16* ap0 = xb + aoff;
        const _Float16* ap1 = ap0 + 16 * XSTR;
        v16h a0 = FRAG16(ap0);
        v16h a1 = FRAG16(ap1);
#pragma unroll
        for (int g = 0; g < 4; ++g) {
            const _Float16* bp = wb + boff + g * (BH * WSTR);
            v16h b = FRAG16(bp);
            acc[g][0] = __builtin_amdgcn_wmma_f32_16x16x32_f16(false, a0, false, b,
                                                               (short)0, acc[g][0], false, false);
            acc[g][1] = __builtin_amdgcn_wmma_f32_16x16x32_f16(false, a1, false, b,
                                                               (short)0, acc[g][1], false, false);
        }
    }

    // ---- epilogue: all 4 gates for each (b,h) live in the same lane/slot ----
    const int train = trainp[0];
    const int h = h0 + waveH * 16 + l16;
    const float bi = biasC[h];
    const float bf = biasC[h + 1024];
    const float bc = biasC[h + 2048];
    const float bo = biasC[h + 3072];
#pragma unroll
    for (int r = 0; r < 2; ++r) {
#pragma unroll
        for (int v = 0; v < 8; ++v) {
            int row = m0 + waveM * 32 + r * 16 + v + half * 8;  // C layout: M = v (+8 hi lanes)
            float ig = quantU8(1.0f / (1.0f + __expf(-(acc[0][r][v] + bi))), train);
            float fg = quantU8(1.0f / (1.0f + __expf(-(acc[1][r][v] + bf))), train);
            float cg = quantS8(tanhf(acc[2][r][v] + bc), train);
            float og = quantU8(1.0f / (1.0f + __expf(-(acc[3][r][v] + bo))), train);
            float cxe = cx[(size_t)row * H_DIM + h];
            float t1 = quantS8(fg * cxe, train);
            float t2 = quantS8(ig * cg, train);
            float cye = quantS8((t1 + t2) * 0.5f, train);
            float hye = quantS8(og * quantS8(tanhf(cye * 2.0f), train), train);
            hy[(size_t)row * H_DIM + h] = hye;
            cy[(size_t)row * H_DIM + h] = cye;
        }
    }
}

extern "C" void kernel_launch(void* const* d_in, const int* in_sizes, int n_in,
                              void* d_out, int out_size, void* d_ws, size_t ws_size,
                              hipStream_t stream) {
    const float* input = (const float*)d_in[0];
    const float* hx    = (const float*)d_in[1];
    const float* cx    = (const float*)d_in[2];
    const float* w_ih  = (const float*)d_in[3];
    const float* w_hh  = (const float*)d_in[4];
    const float* b_ih  = (const float*)d_in[5];
    const float* b_hh  = (const float*)d_in[6];
    const float* nw_ih = (const float*)d_in[7];
    const float* nw_hh = (const float*)d_in[8];
    const float* nb_ih = (const float*)d_in[9];
    const float* nb_hh = (const float*)d_in[10];
    const int*   trainp = (const int*)d_in[11];

    char* ws = (char*)d_ws;
    _Float16* Xq    = (_Float16*)ws;                                   // 16 MB
    _Float16* Weff  = (_Float16*)(ws + (size_t)16 * 1024 * 1024);      // 16 MB
    float*    biasC = (float*)   (ws + (size_t)32 * 1024 * 1024);      // 16 KB
    unsigned* maxes = (unsigned*)(ws + (size_t)32 * 1024 * 1024 + 64 * 1024);

    float* hy = (float*)d_out;
    float* cy = hy + (size_t)B_DIM * H_DIM;

    init_max_kernel<<<1, 32, 0, stream>>>(maxes);
    reduce_max_kernel<<<64, 256, 0, stream>>>(w_ih, NG * I_DIM, maxes + 0);
    reduce_max_kernel<<<64, 256, 0, stream>>>(w_hh, NG * H_DIM, maxes + 1);
    reduce_max_kernel<<<1, 256, 0, stream>>>(b_ih, NG, maxes + 2);
    reduce_max_kernel<<<1, 256, 0, stream>>>(b_hh, NG, maxes + 3);

    bias_prep_kernel<<<NG / 256, 256, 0, stream>>>(b_ih, b_hh, nb_ih, nb_hh, maxes, biasC);
    weight_prep_kernel<<<dim3(K_DIM / 32, NG / 32), dim3(32, 8), 0, stream>>>(
        w_ih, w_hh, nw_ih, nw_hh, maxes, Weff);
    x_prep_kernel<<<(B_DIM * K_DIM) / 256, 256, 0, stream>>>(input, hx, trainp, Xq);

    lstm_gemm_kernel<<<dim3(B_DIM / BM, H_DIM / BH), 256, 0, stream>>>(
        Xq, Weff, biasC, cx, trainp, hy, cy);
}
